// VNAttention_39092792328235
// MI455X (gfx1250) — matrix operand order repaired and checked by reading, not compiled, not run on hardware
//
#include <hip/hip_runtime.h>
#include <hip/hip_bf16.h>

// ---------------------------------------------------------------------------
// VNAttention on MI455X (gfx1250): bf16 WMMA pipeline with async-to-LDS DMA
//   x[2,2048,256,3] --qkv gemm--> q,k,v[16,2048,192] --flash attn--> o --out gemm--> y
// ---------------------------------------------------------------------------

typedef __attribute__((ext_vector_type(16))) __bf16 v16bf;
typedef __attribute__((ext_vector_type(8)))  float  v8f;

union FragBF { v16bf v; unsigned short u[16]; };

#define WMMA_BF16(A, B, C) \
    __builtin_amdgcn_wmma_f32_16x16x32_bf16(false, (A), false, (B), (short)0, (C), false, false)

// One wave-wide async DMA op: each lane moves 16 B global -> LDS (ASYNCcnt).
#define ASYNC_LDS_B128(ldsaddr, gaddr) \
    asm volatile("global_load_async_to_lds_b128 %0, %1, off" \
                 :: "v"(ldsaddr), "v"(gaddr) : "memory")
#define WAIT_ASYNC0() asm volatile("s_wait_asynccnt 0x0" ::: "memory")

static constexpr int   TOK    = 4096;              // b*n
static constexpr int   ROWS   = TOK * 3;           // (tok, coord) rows = 12288
static constexpr int   IFEAT  = 256;
static constexpr int   OQKV   = 1536;
static constexpr int   KD     = 192;               // 3 * 64 attention contraction
static constexpr int   SEQ    = 2048;
static constexpr float SCALE  = 0.07216878364870323f;  // (3*64)^-0.5

// workspace layout (bytes)
static constexpr size_t OFF_XT = 0;
static constexpr size_t SZ_XT  = (size_t)ROWS * IFEAT * 2;        // 6.29 MB
static constexpr size_t OFF_WQ = OFF_XT + SZ_XT;
static constexpr size_t SZ_WQ  = (size_t)OQKV * IFEAT * 2;
static constexpr size_t OFF_WO = OFF_WQ + SZ_WQ;
static constexpr size_t SZ_WO  = (size_t)256 * 512 * 2;
static constexpr size_t OFF_Q  = OFF_WO + SZ_WO;
static constexpr size_t SZ_ATT = (size_t)16 * SEQ * KD * 2;       // 12.58 MB each
static constexpr size_t OFF_K  = OFF_Q + SZ_ATT;
static constexpr size_t OFF_V  = OFF_K + SZ_ATT;
static constexpr size_t OFF_OT = OFF_V + SZ_ATT;                  // Ot[(tok,c), 512] bf16

__device__ __forceinline__ unsigned short f2bf(float f) {
    union { float f; unsigned u; } c; c.f = f;
    unsigned u = c.u;
    unsigned r = u + 0x7FFFu + ((u >> 16) & 1u);  // round-to-nearest-even
    return (unsigned short)(r >> 16);
}

__device__ __forceinline__ unsigned lds_off(const void* p) {
    // generic LDS pointer: low 32 bits are the LDS byte address
    return (unsigned)(unsigned long long)p;
}

// ---------------------------------------------------------------------------
// Kernel 0: fp32 -> bf16 conversion + layout transform for x
// ---------------------------------------------------------------------------
__global__ __launch_bounds__(256) void k_prep(const float* __restrict__ x,
                                              const float* __restrict__ wqkv,
                                              const float* __restrict__ wout,
                                              unsigned short* __restrict__ Xt,
                                              unsigned short* __restrict__ Wqb,
                                              unsigned short* __restrict__ Wob) {
    const int XT_E = ROWS * IFEAT;      // 3,145,728
    const int WQ_E = OQKV * IFEAT;      //   393,216
    const int WO_E = 256 * 512;         //   131,072
    int gid = blockIdx.x * 256 + threadIdx.x;
    if (gid < XT_E) {
        int row = gid >> 8;             // (tok, c)
        int i   = gid & 255;
        int tok = row / 3;
        int c   = row - tok * 3;
        Xt[gid] = f2bf(x[((size_t)tok * IFEAT + i) * 3 + c]);
    } else if (gid < XT_E + WQ_E) {
        int e = gid - XT_E;
        Wqb[e] = f2bf(wqkv[e]);
    } else if (gid < XT_E + WQ_E + WO_E) {
        int e = gid - XT_E - WQ_E;
        Wob[e] = f2bf(wout[e]);
    }
}

// ---------------------------------------------------------------------------
// Shared WMMA GEMM mainloop: block tile 64(M) x 64(N), K-step 32, 4 waves,
// double-buffered async-to-LDS staging overlapped with WMMA.
// ---------------------------------------------------------------------------
__device__ __forceinline__ void gemm_mainloop(const unsigned short* __restrict__ A,
                                              const unsigned short* __restrict__ Bw,
                                              int klen, int m0, int n0, int lda, int ldb,
                                              unsigned short (&Al)[2][64][32],
                                              unsigned short (&Bl)[2][64][32],
                                              v8f (&acc)[4]) {
    const int tid  = threadIdx.x;
    const int w    = tid >> 5;
    const int l    = tid & 31;
    const int l15  = l & 15;
    const int hiK8  = (l & 16) ? 8 : 0;    // A-frag K offset
    const int hiK16 = (l & 16) ? 16 : 0;   // unused; B-frag uses hiK8 too
    const int m  = tid >> 1;               // staged row (0..63)
    const int kh = (tid & 1) * 16;         // staged half-row

    // issue the 4 async b128 ops (2 for A, 2 for B) staging K-step k0 into buf
    auto issue = [&](int buf, int k0) {
        unsigned la = lds_off(&Al[buf][m][kh]);
        unsigned lb = lds_off(&Bl[buf][m][kh]);
        unsigned long long ga = (unsigned long long)(const void*)(A  + (size_t)(m0 + m) * lda + k0 + kh);
        unsigned long long gb = (unsigned long long)(const void*)(Bw + (size_t)(n0 + m) * ldb + k0 + kh);
        ASYNC_LDS_B128(la,        ga);
        ASYNC_LDS_B128(la + 16u,  ga + 16ull);
        ASYNC_LDS_B128(lb,        gb);
        ASYNC_LDS_B128(lb + 16u,  gb + 16ull);
    };

    issue(0, 0);
    WAIT_ASYNC0();
    __syncthreads();

    const int nsteps = klen / 32;
    for (int s = 0; s < nsteps; ++s) {
        const int buf = s & 1;
        if (s + 1 < nsteps) issue(buf ^ 1, (s + 1) * 32);   // DMA next tile now

        FragBF a;
#pragma unroll
        for (int j = 0; j < 16; ++j) {
            int k = (j < 8 ? j : j + 8) + hiK8;
            a.u[j] = Al[buf][w * 16 + l15][k];
        }
#pragma unroll
        for (int nf = 0; nf < 4; ++nf) {
            FragBF b;
#pragma unroll
            for (int j = 0; j < 16; ++j)
                b.u[j] = Bl[buf][nf * 16 + l15][(j < 8 ? j : j + 8) + hiK8];
            acc[nf] = WMMA_BF16(a.v, b.v, acc[nf]);
        }

        WAIT_ASYNC0();       // next tile DMA complete
        __syncthreads();     // all waves done reading buf; LDS visible
    }
}

// ---------------------------------------------------------------------------
// Kernel 1: QKV projection  [12288,256] x [1536,256]^T, scatter into
// attention layout Q/K/V[bh][n][c*64+d] (Q pre-scaled).
// ---------------------------------------------------------------------------
__global__ __launch_bounds__(128) void k_gemm_qkv(const unsigned short* __restrict__ Xt,
                                                  const unsigned short* __restrict__ Wqb,
                                                  unsigned short* __restrict__ Qa,
                                                  unsigned short* __restrict__ Ka,
                                                  unsigned short* __restrict__ Va) {
    __shared__ __align__(16) unsigned short Al[2][64][32];
    __shared__ __align__(16) unsigned short Bl[2][64][32];
    const int m0 = blockIdx.x * 64;
    const int n0 = blockIdx.y * 64;
    v8f acc[4] = {};
    gemm_mainloop(Xt, Wqb, IFEAT, m0, n0, IFEAT, IFEAT, Al, Bl, acc);

    const int tid = threadIdx.x;
    const int w   = tid >> 5;
    const int l   = tid & 31;
    const int l15 = l & 15;
    const int hi8 = (l & 16) ? 8 : 0;
#pragma unroll
    for (int nf = 0; nf < 4; ++nf) {
#pragma unroll
        for (int r = 0; r < 8; ++r) {
            int row = m0 + w * 16 + r + hi8;   // (tok, c)
            int o   = n0 + nf * 16 + l15;
            int tok = row / 3;
            int c   = row - tok * 3;
            int b   = tok >> 11;
            int n   = tok & 2047;
            float val = acc[nf][r];
            int h, d;
            if (o < 512) {
                h = o >> 6; d = o & 63;
                Qa[((size_t)(b * 8 + h) * SEQ + n) * KD + c * 64 + d] = f2bf(val * SCALE);
            } else if (o < 1024) {
                int oo = o - 512; h = oo >> 6; d = oo & 63;
                Ka[((size_t)(b * 8 + h) * SEQ + n) * KD + c * 64 + d] = f2bf(val);
            } else {
                int oo = o - 1024; h = oo >> 6; d = oo & 63;
                Va[((size_t)(b * 8 + h) * SEQ + n) * KD + c * 64 + d] = f2bf(val);
            }
        }
    }
}

// ---------------------------------------------------------------------------
// Kernel 2: fused flash attention. One wave per (bh, 16-query tile).
// Q tile resident in VGPRs (6 A-frags); K/V streamed via async-to-LDS DMA.
// Online softmax over the 2048 keys; O accumulated in 6 f32 C-frags.
// ---------------------------------------------------------------------------
__global__ __launch_bounds__(32) void k_attn(const unsigned short* __restrict__ Qp,
                                             const unsigned short* __restrict__ Kp,
                                             const unsigned short* __restrict__ Vp,
                                             unsigned short* __restrict__ Ot) {
    __shared__ __align__(16) unsigned short Kl[32][KD];
    __shared__ __align__(16) unsigned short Vl[32][KD];
    __shared__ __align__(16) unsigned short Pl[16][32];

    const int bh  = blockIdx.y;
    const int n0  = blockIdx.x * 16;
    const int l   = threadIdx.x;
    const int l15 = l & 15;
    const int hi8  = (l & 16) ? 8 : 0;
    const int hi16 = (l & 16) ? 16 : 0;
    const size_t base = (size_t)bh * SEQ * KD;

    // Q tile -> 6 A fragments (per-lane gather, once)
    FragBF aQ[6];
#pragma unroll
    for (int kk = 0; kk < 6; ++kk)
#pragma unroll
        for (int j = 0; j < 16; ++j) {
            int k = kk * 32 + (j < 8 ? j : j + 8) + hi8;
            aQ[kk].u[j] = Qp[base + (size_t)(n0 + l15) * KD + k];
        }

    v8f cO[6] = {};
    float mx[8], sm[8];
#pragma unroll
    for (int r = 0; r < 8; ++r) { mx[r] = -3.0e38f; sm[r] = 0.0f; }

    const unsigned lK0 = lds_off(&Kl[l][0]);
    const unsigned lV0 = lds_off(&Vl[l][0]);

    for (int j0 = 0; j0 < SEQ; j0 += 32) {
        // async DMA: 32 K rows + 32 V rows (384 B each lane-row) into LDS
        {
            unsigned lK = lK0, lV = lV0;
            unsigned long long gK = (unsigned long long)(const void*)(Kp + base + (size_t)(j0 + l) * KD);
            unsigned long long gV = (unsigned long long)(const void*)(Vp + base + (size_t)(j0 + l) * KD);
#pragma unroll
            for (int q = 0; q < 24; ++q) {
                ASYNC_LDS_B128(lK, gK);
                ASYNC_LDS_B128(lV, gV);
                lK += 16u; lV += 16u; gK += 16ull; gV += 16ull;
            }
        }
        if (j0 + 32 < SEQ) {   // warm L2 for the tile after this one
            __builtin_prefetch(Kp + base + (size_t)(j0 + 32 + l) * KD, 0, 1);
            __builtin_prefetch(Vp + base + (size_t)(j0 + 32 + l) * KD, 0, 1);
        }
        WAIT_ASYNC0();
        __syncthreads();

#pragma unroll
        for (int sub = 0; sub < 2; ++sub) {
            // S = Q (16x192) . K^T (192x16) over this 16-key subtile
            v8f cS = {};
#pragma unroll
            for (int kk = 0; kk < 6; ++kk) {
                FragBF bK;
#pragma unroll
                for (int j = 0; j < 16; ++j)
                    bK.u[j] = Kl[sub * 16 + l15][kk * 32 + (j < 8 ? j : j + 8) + hi8];
                cS = WMMA_BF16(aQ[kk].v, bK.v, cS);
            }
            // online softmax: row stats live per half-wave (16 lanes = 16 cols)
#pragma unroll
            for (int r = 0; r < 8; ++r) {
                float v  = cS[r];
                float rm = v;
                rm = fmaxf(rm, __shfl_xor(rm, 1, 32));
                rm = fmaxf(rm, __shfl_xor(rm, 2, 32));
                rm = fmaxf(rm, __shfl_xor(rm, 4, 32));
                rm = fmaxf(rm, __shfl_xor(rm, 8, 32));
                float mnew = fmaxf(mx[r], rm);
                float corr = __expf(mx[r] - mnew);
                mx[r] = mnew;
                float p  = __expf(v - mnew);
                float rs = p;
                rs += __shfl_xor(rs, 1, 32);
                rs += __shfl_xor(rs, 2, 32);
                rs += __shfl_xor(rs, 4, 32);
                rs += __shfl_xor(rs, 8, 32);
                sm[r] = sm[r] * corr + rs;
#pragma unroll
                for (int t = 0; t < 6; ++t) cO[t][r] = cO[t][r] * corr;
                Pl[r + hi8][sub * 16 + l15] = f2bf(p);   // C-layout -> LDS
            }
        }
        __syncthreads();

        // O += P (16x32) . V (32x192)
        FragBF aP;
#pragma unroll
        for (int j = 0; j < 16; ++j)
            aP.u[j] = Pl[l15][(j < 8 ? j : j + 8) + hi8];
#pragma unroll
        for (int t = 0; t < 6; ++t) {
            FragBF bV;
#pragma unroll
            for (int j = 0; j < 16; ++j)
                bV.u[j] = Vl[(j < 8 ? j : j + 8) + hi8][t * 16 + l15];
            cO[t] = WMMA_BF16(aP.v, bV.v, cO[t]);
        }
        __syncthreads();
    }

    // normalize and scatter to Ot[(tok,c), h*64+d]
    const int b = bh >> 3, h = bh & 7;
#pragma unroll
    for (int r = 0; r < 8; ++r) {
        float inv = 1.0f / sm[r];
        int n = n0 + r + hi8;
        size_t tok = (size_t)b * SEQ + n;
#pragma unroll
        for (int t = 0; t < 6; ++t) {
            int kidx = t * 16 + l15;
            int c = kidx >> 6, d = kidx & 63;
            Ot[(tok * 3 + c) * 512 + h * 64 + d] = f2bf(cO[t][r] * inv);
        }
    }
}

// ---------------------------------------------------------------------------
// Kernel 3: output projection  [12288,512] x [256,512]^T -> y[b,n,256,3] fp32
// ---------------------------------------------------------------------------
__global__ __launch_bounds__(128) void k_out(const unsigned short* __restrict__ Ot,
                                             const unsigned short* __restrict__ Wob,
                                             float* __restrict__ out) {
    __shared__ __align__(16) unsigned short Al[2][64][32];
    __shared__ __align__(16) unsigned short Bl[2][64][32];
    const int m0 = blockIdx.x * 64;
    const int n0 = blockIdx.y * 64;
    v8f acc[4] = {};
    gemm_mainloop(Ot, Wob, 512, m0, n0, 512, 512, Al, Bl, acc);

    const int tid = threadIdx.x;
    const int w   = tid >> 5;
    const int l   = tid & 31;
    const int l15 = l & 15;
    const int hi8 = (l & 16) ? 8 : 0;
#pragma unroll
    for (int nf = 0; nf < 4; ++nf) {
#pragma unroll
        for (int r = 0; r < 8; ++r) {
            int row = m0 + w * 16 + r + hi8;
            int o   = n0 + nf * 16 + l15;
            int tok = row / 3;
            int c   = row - tok * 3;
            out[((size_t)tok * 256 + o) * 3 + c] = acc[nf][r];
        }
    }
}

// ---------------------------------------------------------------------------
extern "C" void kernel_launch(void* const* d_in, const int* in_sizes, int n_in,
                              void* d_out, int out_size, void* d_ws, size_t ws_size,
                              hipStream_t stream) {
    const float* x    = (const float*)d_in[0];
    const float* wqkv = (const float*)d_in[1];
    const float* wout = (const float*)d_in[2];
    float* out = (float*)d_out;

    char* ws = (char*)d_ws;
    unsigned short* Xt  = (unsigned short*)(ws + OFF_XT);
    unsigned short* Wqb = (unsigned short*)(ws + OFF_WQ);
    unsigned short* Wob = (unsigned short*)(ws + OFF_WO);
    unsigned short* Qa  = (unsigned short*)(ws + OFF_Q);
    unsigned short* Ka  = (unsigned short*)(ws + OFF_K);
    unsigned short* Va  = (unsigned short*)(ws + OFF_V);
    unsigned short* Ot  = (unsigned short*)(ws + OFF_OT);

    // 0) precision/layout prep
    {
        int total = ROWS * IFEAT + OQKV * IFEAT + 256 * 512;
        int blocks = (total + 255) / 256;
        k_prep<<<blocks, 256, 0, stream>>>(x, wqkv, wout, Xt, Wqb, Wob);
    }
    // 1) QKV projection (WMMA)
    {
        dim3 grid(ROWS / 64, OQKV / 64);   // 192 x 24
        k_gemm_qkv<<<grid, 128, 0, stream>>>(Xt, Wqb, Qa, Ka, Va);
    }
    // 2) fused flash attention (WMMA, async-to-LDS K/V stream)
    {
        dim3 grid(SEQ / 16, 16);           // 128 x 16
        k_attn<<<grid, 32, 0, stream>>>(Qa, Ka, Va, Ot);
    }
    // 3) output projection (WMMA)
    {
        dim3 grid(ROWS / 64, 256 / 64);    // 192 x 4
        k_out<<<grid, 128, 0, stream>>>(Ot, Wob, out);
    }
}
